// LocalCostVolumeEncoder_45715631899341
// MI455X (gfx1250) — matrix operand order repaired and checked by reading, not compiled, not run on hardware
//
#include <hip/hip_runtime.h>
#include <stdint.h>

// ---------------------------------------------------------------------------
// Problem constants (B=1)
// ---------------------------------------------------------------------------
#define NVOX   110592      // 48*48*48
#define DIMW   48
#define DIMHW  2304        // 48*48
#define CIN    64
#define PC     64
#define OC     128
#define KOFF   343         // 7^3 offsets
#define CPT1   11          // 352/32 channel groups (enc1: 343 corr + 3 eo + 6 pad)
#define CPT2   4           // 128/32 channel groups (enc2)

typedef __attribute__((ext_vector_type(16))) __bf16 v16bf;
typedef __attribute__((ext_vector_type(8)))  float  v8f;

union FragBF { v16bf v; uint4 u[2]; };

static __device__ __forceinline__ v8f zero8() {
    v8f z;
#pragma unroll
    for (int i = 0; i < 8; ++i) z[i] = 0.0f;
    return z;
}

static __device__ __forceinline__ uint16_t f32_to_bf16(float f) {
    uint32_t x = __float_as_uint(f);
    x += 0x7FFFu + ((x >> 16) & 1u);          // round-to-nearest-even
    return (uint16_t)(x >> 16);
}
static __device__ __forceinline__ float bf16lo(uint32_t u) { return __uint_as_float(u << 16); }
static __device__ __forceinline__ float bf16hi(uint32_t u) { return __uint_as_float(u & 0xFFFF0000u); }

// ---------------------------------------------------------------------------
// Kernel 1: 1x1x1 projection (64->64) + bias + channel L2 norm -> bf16 [n][64]
// grid: (432, 2)  block: 256.  blockIdx.y: 0=source, 1=target
// ---------------------------------------------------------------------------
__global__ __launch_bounds__(256)
void proj_kernel(const float* __restrict__ xsrc, const float* __restrict__ xtgt,
                 const float* __restrict__ wsrc, const float* __restrict__ bsrc,
                 const float* __restrict__ wtgt, const float* __restrict__ btgt,
                 uint16_t* __restrict__ spo, uint16_t* __restrict__ tpo)
{
    const float* xin; const float* wct; const float* bct; uint16_t* outp;
    if (blockIdx.y == 0) { xin = xsrc; wct = wsrc; bct = bsrc; outp = spo; }
    else                 { xin = xtgt; wct = wtgt; bct = btgt; outp = tpo; }

    __shared__ float wl[PC * CIN];
    __shared__ float bl[PC];
    for (int i = threadIdx.x; i < PC * CIN; i += 256) wl[i] = wct[i];
    if (threadIdx.x < PC) bl[threadIdx.x] = bct[threadIdx.x];
    __syncthreads();

    const int n = blockIdx.x * 256 + threadIdx.x;

    float xv[CIN];
#pragma unroll
    for (int c = 0; c < CIN; ++c) xv[c] = xin[(size_t)c * NVOX + n];

    // pass 1: squared norm
    float sq = 0.0f;
    for (int pc = 0; pc < PC; ++pc) {
        float s = bl[pc];
#pragma unroll
        for (int c = 0; c < CIN; ++c) s = fmaf(wl[pc * CIN + c], xv[c], s);
        sq = fmaf(s, s, sq);
    }
    const float rn = 1.0f / fmaxf(sqrtf(sq), 1e-12f);

    // pass 2: recompute, normalize, pack bf16 pairs
    uint32_t* op = (uint32_t*)(outp + (size_t)n * PC);
    for (int pc = 0; pc < PC; pc += 2) {
        float s0 = bl[pc], s1 = bl[pc + 1];
#pragma unroll
        for (int c = 0; c < CIN; ++c) {
            s0 = fmaf(wl[pc * CIN + c],       xv[c], s0);
            s1 = fmaf(wl[(pc + 1) * CIN + c], xv[c], s1);
        }
        op[pc >> 1] = (uint32_t)f32_to_bf16(s0 * rn) | ((uint32_t)f32_to_bf16(s1 * rn) << 16);
    }
}

// ---------------------------------------------------------------------------
// Kernel 2: local cost volume (343 clamped shifts) + softmax expected offset.
// feat1 grouped layout: [g=0..10][voxel][32 bf16], g*32+c channel order:
//   0..342 = corr, 343..345 = expected offset (z,y,x), 346..351 = zero pad
// Each voxel's 32-channel group is 64B contiguous -> coalesced b128 stores and
// async-LDS-friendly gathers in the conv kernel.
// grid: 432  block: 256 (thread per voxel)
// ---------------------------------------------------------------------------
__global__ __launch_bounds__(256)
void corr_kernel(const uint16_t* __restrict__ sp, const uint16_t* __restrict__ tp,
                 uint16_t* __restrict__ feat1)
{
    const int n = blockIdx.x * 256 + threadIdx.x;
    const int z = n / DIMHW, y = (n % DIMHW) / DIMW, x = n % DIMW;

    uint4 s4[8];
    const uint4* spv = (const uint4*)(sp + (size_t)n * PC);
#pragma unroll
    for (int i = 0; i < 8; ++i) s4[i] = spv[i];

    uint32_t buf[16];
    float ssum = 0.f, vz = 0.f, vy = 0.f, vx = 0.f;
    for (int tap = 0; tap < KOFF; ++tap) {
        const int dz = tap / 49 - 3;
        const int rem = tap % 49;
        const int dy = rem / 7 - 3;
        const int dx = rem % 7 - 3;
        const int zz = min(max(z + dz, 0), 47);   // edge padding == clamp
        const int yy = min(max(y + dy, 0), 47);
        const int xx = min(max(x + dx, 0), 47);
        const uint4* tv = (const uint4*)(tp + (size_t)((zz * DIMW + yy) * DIMW + xx) * PC);
        float c = 0.0f;
#pragma unroll
        for (int i = 0; i < 8; ++i) {
            const uint4 b4 = tv[i];
            c = fmaf(bf16lo(s4[i].x), bf16lo(b4.x), c);
            c = fmaf(bf16hi(s4[i].x), bf16hi(b4.x), c);
            c = fmaf(bf16lo(s4[i].y), bf16lo(b4.y), c);
            c = fmaf(bf16hi(s4[i].y), bf16hi(b4.y), c);
            c = fmaf(bf16lo(s4[i].z), bf16lo(b4.z), c);
            c = fmaf(bf16hi(s4[i].z), bf16hi(b4.z), c);
            c = fmaf(bf16lo(s4[i].w), bf16lo(b4.w), c);
            c = fmaf(bf16hi(s4[i].w), bf16hi(b4.w), c);
        }
        const int slot = tap & 31;
        const uint32_t h = f32_to_bf16(c);
        if ((slot & 1) == 0) buf[slot >> 1] = h;
        else                 buf[slot >> 1] |= h << 16;

        // |c| <= 1 after L2 norm -> plain exp is safe (no max-tracking needed)
        const float e = __expf(c);
        ssum += e;
        vz = fmaf(e, (float)dz, vz);
        vy = fmaf(e, (float)dy, vy);
        vx = fmaf(e, (float)dx, vx);

        if (slot == 31) {
            uint4* dst = (uint4*)(feat1 + ((size_t)(tap >> 5) * NVOX + n) * 32);
#pragma unroll
            for (int q = 0; q < 4; ++q)
                dst[q] = make_uint4(buf[q * 4], buf[q * 4 + 1], buf[q * 4 + 2], buf[q * 4 + 3]);
        }
    }
    // group 10: slots 0..22 = taps 320..342 (already in buf), 23..25 = eo, 26..31 = 0
    const float inv = 1.0f / ssum;
    buf[11] = (buf[11] & 0xFFFFu) | ((uint32_t)f32_to_bf16(vz * inv) << 16);
    buf[12] = (uint32_t)f32_to_bf16(vy * inv) | ((uint32_t)f32_to_bf16(vx * inv) << 16);
    buf[13] = 0; buf[14] = 0; buf[15] = 0;
    uint4* dst = (uint4*)(feat1 + ((size_t)10 * NVOX + n) * 32);
#pragma unroll
    for (int q = 0; q < 4; ++q)
        dst[q] = make_uint4(buf[q * 4], buf[q * 4 + 1], buf[q * 4 + 2], buf[q * 4 + 3]);
}

// ---------------------------------------------------------------------------
// Kernel 3: weight repack -> bf16, exact WMMA A-fragment byte order.
// K ordering: k = tap*(32*CPT) + ci ; per-lane A k order used here (contiguous, same as B):
//   lanes 0-15 : K{0..15},  lanes 16-31 : K{16..31}
// wpk[((mt*TOTCH + kc)*32 + lane)*16 + e]
// ---------------------------------------------------------------------------
template<int CINP, int CPT>
__global__ void repack_w(const float* __restrict__ w, uint16_t* __restrict__ wpk, int total)
{
    const int idx = blockIdx.x * 256 + threadIdx.x;
    if (idx >= total) return;
    const int e    = idx & 15;
    const int lane = (idx >> 4) & 31;
    const int kc   = (idx >> 9) % (27 * CPT);
    const int mt   = (idx >> 9) / (27 * CPT);
    const int m    = mt * 16 + (lane & 15);
    const int kk   = e + ((lane >= 16) ? 16 : 0);
    const int tap  = kc / CPT;
    const int ci   = (kc % CPT) * 32 + kk;
    const float val = (ci < CINP) ? w[((size_t)m * CINP + ci) * 27 + tap] : 0.0f;
    wpk[idx] = f32_to_bf16(val);
}

// ---------------------------------------------------------------------------
// Kernel 4: 3x3x3 conv (zero pad) as implicit GEMM, v_wmma_f32_16x16x32_bf16.
// Block: 256 thr (8 waves), M=128 out-ch x N=128 voxels. Wave (mtil2,nquad):
// 2 M-tiles x 4 N-tiles = 8 WMMA per K-chunk. B tile staged to LDS with
// global_load_async_to_lds_b32 (IOFFSET advances global + LDS in lockstep),
// zeros written directly for out-of-bounds taps. No bias (cancels under
// affine-free InstanceNorm).
// grid: 864  block: 256
// ---------------------------------------------------------------------------
template<int CPT>
__global__ __launch_bounds__(256)
void conv3x3_wmma(const uint16_t* __restrict__ feat,   // [CPT][NVOX][32] bf16
                  const uint16_t* __restrict__ wpk,    // fragment-packed weights
                  float* __restrict__ yout)            // [128][NVOX] f32
{
    constexpr int TOTCH = 27 * CPT;
    __shared__ __align__(16) char lds8[128 * 80];      // 128 rows x (64B data + 16B pad)

    const int tid = threadIdx.x;
    const int n0  = blockIdx.x * 128;
    const int nl  = tid & 127;                         // staging voxel
    const int pg  = tid >> 7;                          // staging half (16 ch each)
    const int n   = n0 + nl;
    const int z = n / DIMHW, y = (n % DIMHW) / DIMW, x = n % DIMW;

    const int wave  = tid >> 5;
    const int lane  = tid & 31;
    const int mtil2 = wave & 3;                        // M rows [mtil2*32, +32)
    const int nquad = wave >> 2;                       // N rows [nquad*64, +64)
    const int l15   = lane & 15;
    const int kbByte = (lane >> 4) * 32;               // B frag: lanes>=16 hold K 16..31

    // LDS byte address (low 32 bits of generic shared pointer == LDS offset)
    const uint32_t lbase = (uint32_t)(uintptr_t)lds8 + (uint32_t)(nl * 80 + pg * 32);

    v8f acc[2][4];
#pragma unroll
    for (int i = 0; i < 2; ++i)
#pragma unroll
        for (int j = 0; j < 4; ++j) acc[i][j] = zero8();

    for (int tap = 0; tap < 27; ++tap) {
        const int dz = tap / 9 - 1, dy = (tap / 3) % 3 - 1, dx = tap % 3 - 1;
        const int zz = z + dz, yy = y + dy, xx = x + dx;
        const bool valid = (unsigned)zz < 48u && (unsigned)yy < 48u && (unsigned)xx < 48u;
        const int nsh = (zz * DIMW + yy) * DIMW + xx;

        for (int cc = 0; cc < CPT; ++cc) {
            const int chunk = tap * CPT + cc;
            __syncthreads();                           // previous chunk consumed
            // ---- stage B tile [voxel][32k] via async gather into LDS ----
            if (valid) {
                const uint16_t* gp = feat + (((size_t)cc * NVOX + nsh) * 32 + pg * 16);
                asm volatile(
                    "global_load_async_to_lds_b32 %0, %1, off\n\t"
                    "global_load_async_to_lds_b32 %0, %1, off offset:4\n\t"
                    "global_load_async_to_lds_b32 %0, %1, off offset:8\n\t"
                    "global_load_async_to_lds_b32 %0, %1, off offset:12\n\t"
                    "global_load_async_to_lds_b32 %0, %1, off offset:16\n\t"
                    "global_load_async_to_lds_b32 %0, %1, off offset:20\n\t"
                    "global_load_async_to_lds_b32 %0, %1, off offset:24\n\t"
                    "global_load_async_to_lds_b32 %0, %1, off offset:28"
                    :: "v"(lbase), "v"(gp) : "memory");
            } else {
                const uint4 z4 = make_uint4(0u, 0u, 0u, 0u);
                *(uint4*)(lds8 + nl * 80 + pg * 32)      = z4;
                *(uint4*)(lds8 + nl * 80 + pg * 32 + 16) = z4;
            }
            asm volatile("s_wait_asynccnt 0x0" ::: "memory");
            __syncthreads();

            // ---- A fragments: contiguous coalesced 32B per lane ----
            FragBF a[2];
#pragma unroll
            for (int mt = 0; mt < 2; ++mt) {
                const int MT = mtil2 * 2 + mt;
                const uint4* p = (const uint4*)(wpk +
                    ((size_t)(MT * TOTCH + chunk) * 32 + lane) * 16);
                a[mt].u[0] = p[0]; a[mt].u[1] = p[1];
            }
            // ---- B fragments from LDS: 2x ds_load_b128 per tile ----
            FragBF b[4];
#pragma unroll
            for (int nt = 0; nt < 4; ++nt) {
                const int row = nquad * 64 + nt * 16 + l15;
                const uint4* p = (const uint4*)(lds8 + row * 80 + kbByte);
                b[nt].u[0] = p[0]; b[nt].u[1] = p[1];
            }
#pragma unroll
            for (int mt = 0; mt < 2; ++mt)
#pragma unroll
                for (int nt = 0; nt < 4; ++nt)
                    acc[mt][nt] = __builtin_amdgcn_wmma_f32_16x16x32_bf16(
                        false, a[mt].v, false, b[nt].v,
                        (short)0, acc[mt][nt], false, false);
        }
    }

    // ---- store C tiles: VGPR r -> row r + (lane>=16 ? 8 : 0), col = lane&15 ----
#pragma unroll
    for (int mt = 0; mt < 2; ++mt) {
        const int MT = mtil2 * 2 + mt;
#pragma unroll
        for (int nt = 0; nt < 4; ++nt) {
            const int col = n0 + (nquad * 64 + nt * 16) + l15;
#pragma unroll
            for (int r = 0; r < 8; ++r) {
                const int m = MT * 16 + r + ((lane >> 4) << 3);
                yout[(size_t)m * NVOX + col] = acc[mt][nt][r];
            }
        }
    }
}

// ---------------------------------------------------------------------------
// Kernel 5: per-channel mean / rstd (biased var, eps=1e-5). One block/channel.
// ---------------------------------------------------------------------------
__global__ __launch_bounds__(256)
void stats_kernel(const float* __restrict__ ybuf, float* __restrict__ stats)
{
    const int ch = blockIdx.x;
    const float* p = ybuf + (size_t)ch * NVOX;
    float s = 0.f, s2 = 0.f;
    for (int i = threadIdx.x; i < NVOX; i += 256) {
        const float v = p[i];
        s += v; s2 = fmaf(v, v, s2);
    }
    __shared__ float rs[256], rs2[256];
    rs[threadIdx.x] = s; rs2[threadIdx.x] = s2;
    __syncthreads();
    for (int off = 128; off > 0; off >>= 1) {
        if (threadIdx.x < off) {
            rs[threadIdx.x]  += rs[threadIdx.x + off];
            rs2[threadIdx.x] += rs2[threadIdx.x + off];
        }
        __syncthreads();
    }
    if (threadIdx.x == 0) {
        const float mean = rs[0] / (float)NVOX;
        const float var  = rs2[0] / (float)NVOX - mean * mean;
        stats[ch * 2]     = mean;
        stats[ch * 2 + 1] = rsqrtf(var + 1e-5f);
    }
}

// ---------------------------------------------------------------------------
// Kernel 6a: normalize + exact GELU -> bf16, grouped [g][voxel][32] layout.
// grid: (432, 4)  block: 256
// ---------------------------------------------------------------------------
__global__ __launch_bounds__(256)
void norm_gelu_pack(const float* __restrict__ ybuf, const float* __restrict__ stats,
                    uint16_t* __restrict__ feato)
{
    const int n = blockIdx.x * 256 + threadIdx.x;
    const int g = blockIdx.y;
    uint32_t buf[16];
#pragma unroll
    for (int j = 0; j < 32; j += 2) {
        const int c0 = g * 32 + j;
        const float v0 = (ybuf[(size_t)c0 * NVOX + n] - stats[c0 * 2]) * stats[c0 * 2 + 1];
        const float v1 = (ybuf[(size_t)(c0 + 1) * NVOX + n] - stats[(c0 + 1) * 2]) * stats[(c0 + 1) * 2 + 1];
        const float g0 = 0.5f * v0 * (1.0f + erff(v0 * 0.70710678118654752f));
        const float g1 = 0.5f * v1 * (1.0f + erff(v1 * 0.70710678118654752f));
        buf[j >> 1] = (uint32_t)f32_to_bf16(g0) | ((uint32_t)f32_to_bf16(g1) << 16);
    }
    uint4* dst = (uint4*)(feato + ((size_t)g * NVOX + n) * 32);
#pragma unroll
    for (int q = 0; q < 4; ++q)
        dst[q] = make_uint4(buf[q * 4], buf[q * 4 + 1], buf[q * 4 + 2], buf[q * 4 + 3]);
}

// ---------------------------------------------------------------------------
// Kernel 6b: normalize + exact GELU -> f32 final output [128][NVOX]
// ---------------------------------------------------------------------------
__global__ __launch_bounds__(256)
void norm_gelu_out(const float* __restrict__ ybuf, const float* __restrict__ stats,
                   float* __restrict__ outf)
{
    const size_t i = (size_t)blockIdx.x * 256 + threadIdx.x;
    const int ch = (int)(i / NVOX);
    const float v = (ybuf[i] - stats[ch * 2]) * stats[ch * 2 + 1];
    outf[i] = 0.5f * v * (1.0f + erff(v * 0.70710678118654752f));
}

// ---------------------------------------------------------------------------
// Workspace layout (bytes)
// ---------------------------------------------------------------------------
#define OFF_SP     ((size_t)0)                                  // 14,155,776
#define OFF_TP     (OFF_SP + (size_t)NVOX * PC * 2)             // 14,155,776
#define OFF_FEAT1  (OFF_TP + (size_t)NVOX * PC * 2)             // 77,856,768 (feat2 aliases)
#define OFF_WPK1   (OFF_FEAT1 + (size_t)CPT1 * NVOX * 64)       // 2,433,024
#define OFF_WPK2   (OFF_WPK1 + (size_t)8 * 27 * CPT1 * 512 * 2) // 884,736
#define OFF_Y      (OFF_WPK2 + (size_t)8 * 27 * CPT2 * 512 * 2) // 56,623,104
#define OFF_STATS  (OFF_Y + (size_t)OC * NVOX * 4)              // 1,024

extern "C" void kernel_launch(void* const* d_in, const int* in_sizes, int n_in,
                              void* d_out, int out_size, void* d_ws, size_t ws_size,
                              hipStream_t stream) {
    (void)in_sizes; (void)n_in; (void)out_size; (void)ws_size;

    const float* src = (const float*)d_in[0];
    const float* tgt = (const float*)d_in[1];
    const float* psw = (const float*)d_in[2];
    const float* psb = (const float*)d_in[3];
    const float* ptw = (const float*)d_in[4];
    const float* ptb = (const float*)d_in[5];
    const float* w1  = (const float*)d_in[6];
    const float* w2  = (const float*)d_in[8];
    // enc1_b / enc2_b (d_in[7], d_in[9]) cancel under affine-free InstanceNorm.

    char* ws = (char*)d_ws;
    uint16_t* sp    = (uint16_t*)(ws + OFF_SP);
    uint16_t* tp    = (uint16_t*)(ws + OFF_TP);
    uint16_t* feat1 = (uint16_t*)(ws + OFF_FEAT1);
    uint16_t* feat2 = (uint16_t*)(ws + OFF_FEAT1);   // alias: feat1 dead after conv1
    uint16_t* wpk1  = (uint16_t*)(ws + OFF_WPK1);
    uint16_t* wpk2  = (uint16_t*)(ws + OFF_WPK2);
    float*    ybuf  = (float*)(ws + OFF_Y);
    float*    stats = (float*)(ws + OFF_STATS);
    float*    out   = (float*)d_out;

    // projections (src + tgt) and weight repacks (independent)
    proj_kernel<<<dim3(NVOX / 256, 2), 256, 0, stream>>>(src, tgt, psw, psb, ptw, ptb, sp, tp);
    {
        const int t1 = 8 * 27 * CPT1 * 512;
        repack_w<KOFF + 3, CPT1><<<(t1 + 255) / 256, 256, 0, stream>>>(w1, wpk1, t1);
        const int t2 = 8 * 27 * CPT2 * 512;
        repack_w<OC, CPT2><<<(t2 + 255) / 256, 256, 0, stream>>>(w2, wpk2, t2);
    }
    // cost volume + expected offset (grouped bf16 layout)
    corr_kernel<<<NVOX / 256, 256, 0, stream>>>(sp, tp, feat1);
    // enc1
    conv3x3_wmma<CPT1><<<NVOX / 128, 256, 0, stream>>>(feat1, wpk1, ybuf);
    stats_kernel<<<OC, 256, 0, stream>>>(ybuf, stats);
    norm_gelu_pack<<<dim3(NVOX / 256, 4), 256, 0, stream>>>(ybuf, stats, feat2);
    // enc2
    conv3x3_wmma<CPT2><<<NVOX / 128, 256, 0, stream>>>(feat2, wpk2, ybuf);
    stats_kernel<<<OC, 256, 0, stream>>>(ybuf, stats);
    norm_gelu_out<<<(OC * NVOX) / 256, 256, 0, stream>>>(ybuf, stats, out);
}
